// VSSM_Fusion_78219944395336
// MI455X (gfx1250) — matrix operand order repaired and checked by reading, not compiled, not run on hardware
//
#include <hip/hip_runtime.h>

typedef __attribute__((ext_vector_type(16))) _Float16 v16h;
typedef __attribute__((ext_vector_type(8)))  _Float16 v8h;
typedef __attribute__((ext_vector_type(8)))  float    v8f;

static __device__ __forceinline__ float sigmoidf_(float x){ return 1.0f/(1.0f+__expf(-x)); }
static __device__ __forceinline__ float siluf_(float x){ return x*sigmoidf_(x); }
static __device__ __forceinline__ float softplusf_(float x){ return (x>20.f)? x : log1pf(__expf(x)); }

#define SHUF16(a,b) __builtin_shufflevector(a, b, 0,1,2,3,4,5,6,7,8,9,10,11,12,13,14,15)

// ---------------------------------------------------------------------------
// General strided, batched GEMM on the WMMA pipe.
//   C[z][m,n] = sum_k A[za][m*sAm + k*sAk] * B[zb][k*sBk + n*sBn]  (+bias[n], act)
//   za = aZmod ? z % aZmod : z   (batch offsets sAz/sBz/sCz)
// Block = 256 threads = 8 waves; block tile 64x64; each wave owns two 16x16
// M-sub-tiles of one 16-wide N column -> 2 WMMAs per staged K-tile.
// A tile staged row-major As[m][k]; B tile staged TRANSPOSED BsT[n][k] so that
// both fragments are contiguous 16B runs per lane (ds_load_b128), per the
// CDNA5 ISA 7.12.2 16-bit A/B lane layouts. Rows padded to 40 halves (80 B)
// for bank spread while keeping 16B alignment.
// ---------------------------------------------------------------------------
__global__ __launch_bounds__(256) void k_gemm(
    const float* __restrict__ A, const float* __restrict__ B, float* __restrict__ C,
    int M, int N, int K,
    int sAm, int sAk, int sBk, int sBn, int sCm,
    long sAz, long sBz, long sCz, int aZmod, int bZmod,
    const float* __restrict__ bias, int act)
{
  __shared__ __align__(16) _Float16 As [64][40];   // [m][k]
  __shared__ __align__(16) _Float16 BsT[64][40];   // [n][k]
  int z = blockIdx.z;
  const float* Ap = A + (long)(aZmod ? (z % aZmod) : z) * sAz;
  const float* Bp = B + (long)(bZmod ? (z % bZmod) : z) * sBz;
  float* Cp = C + (long)z * sCz;

  int tid  = threadIdx.x;
  int lane = tid & 31;
  int wave = tid >> 5;     // 0..7
  int wrow = wave >> 2;    // 0..1 -> m sub-tiles at wrow*16 and wrow*16+32
  int wcol = wave & 3;     // 0..3 -> n tile at wcol*16
  int half = lane >> 4;    // 0..1
  int mr   = lane & 15;

  int m0 = blockIdx.x * 64;
  int n0 = blockIdx.y * 64;

  // uniform vectorizable-path predicates (16B-aligned float4 loads along k)
  bool aVec = (sAk == 1) && ((((unsigned long)(const void*)Ap) & 15ul) == 0) && ((sAm & 3) == 0);
  bool bVec = (sBk == 1) && ((((unsigned long)(const void*)Bp) & 15ul) == 0) && ((sBn & 3) == 0);

  v8f acc0 = {0.f,0.f,0.f,0.f,0.f,0.f,0.f,0.f};
  v8f acc1 = {0.f,0.f,0.f,0.f,0.f,0.f,0.f,0.f};

  for (int kk = 0; kk < K; kk += 32) {
    bool fullK = (kk + 32 <= K);

    // ---- stage A tile (64 x 32) ----
    if (aVec && fullK && (m0 + 64 <= M)) {
      int r = tid >> 2;            // 0..63
      int c = (tid & 3) * 8;       // 0,8,16,24
      const float* src = Ap + (long)(m0 + r) * sAm + (kk + c);
      float4 f0 = ((const float4*)src)[0];
      float4 f1 = ((const float4*)src)[1];
      v8h hv = { (_Float16)f0.x,(_Float16)f0.y,(_Float16)f0.z,(_Float16)f0.w,
                 (_Float16)f1.x,(_Float16)f1.y,(_Float16)f1.z,(_Float16)f1.w };
      *(v8h*)&As[r][c] = hv;
      if (kk + 32 < K) __builtin_prefetch(src + 32, 0, 1);
    } else {
#pragma unroll
      for (int it = 0; it < 8; ++it) {
        int t = tid + it*256;
        int r = t >> 5, c = t & 31;
        int gm = m0 + r, gk = kk + c;
        // clamped unconditional load + select: keeps loads pipelined
        int cm = min(gm, M-1), ck = min(gk, K-1);
        float v = Ap[(long)cm*sAm + (long)ck*sAk];
        As[r][c] = (_Float16)((gm < M && gk < K) ? v : 0.f);
      }
    }

    // ---- stage B tile transposed: BsT[n][k] (64 x 32) ----
    if (bVec && fullK && (n0 + 64 <= N)) {
      int nq = tid >> 2;
      int c  = (tid & 3) * 8;
      const float* src = Bp + (long)(n0 + nq) * sBn + (kk + c);
      float4 f0 = ((const float4*)src)[0];
      float4 f1 = ((const float4*)src)[1];
      v8h hv = { (_Float16)f0.x,(_Float16)f0.y,(_Float16)f0.z,(_Float16)f0.w,
                 (_Float16)f1.x,(_Float16)f1.y,(_Float16)f1.z,(_Float16)f1.w };
      *(v8h*)&BsT[nq][c] = hv;
      if (kk + 32 < K) __builtin_prefetch(src + 32, 0, 1);
    } else {
#pragma unroll
      for (int it = 0; it < 8; ++it) {
        int t = tid + it*256;
        int kq = t & 31, nq = t >> 5;
        int gk = kk + kq, gn = n0 + nq;
        int ck = min(gk, K-1), cn = min(gn, N-1);
        float v = Bp[(long)ck*sBk + (long)cn*sBn];
        BsT[nq][kq] = (_Float16)((gk < K && gn < N) ? v : 0.f);
      }
    }
    __syncthreads();

    // ---- fragments: contiguous 16B runs per lane ----
    const _Float16* ar0 = &As[wrow*16      + mr][0];
    const _Float16* ar1 = &As[wrow*16 + 32 + mr][0];
    const _Float16* br  = &BsT[wcol*16 + mr][0];
    // A 16x32 f16 layout: e<8 -> K = 8*half+e ; e>=8 -> K = 16+8*half+(e-8)
    v8h a00 = *(const v8h*)(ar0 + 8*half);
    v8h a01 = *(const v8h*)(ar0 + 16 + 8*half);
    v8h a10 = *(const v8h*)(ar1 + 8*half);
    v8h a11 = *(const v8h*)(ar1 + 16 + 8*half);
    // B 32x16 f16 layout: e<8 -> K = 8*half+e ; e>=8 -> K = 16+8*half+(e-8)
    v8h b0  = *(const v8h*)(br + 8*half);
    v8h b1  = *(const v8h*)(br + 16 + 8*half);
    v16h af0 = SHUF16(a00, a01);
    v16h af1 = SHUF16(a10, a11);
    v16h bf  = SHUF16(b0,  b1);
    acc0 = __builtin_amdgcn_wmma_f32_16x16x32_f16(false, af0, false, bf, (short)0, acc0, false, false);
    acc1 = __builtin_amdgcn_wmma_f32_16x16x32_f16(false, af1, false, bf, (short)0, acc1, false, false);
    __syncthreads();
  }

#pragma unroll
  for (int r = 0; r < 8; ++r) {
    // f32 C 16x16 layout: VGPR r -> M = r + 8*half, N = mr
    int gn = n0 + wcol*16 + mr;
    int gm = m0 + wrow*16 + r + 8*half;
    if (gm < M && gn < N) {
      float v = acc0[r];
      if (bias) v += bias[gn];
      if (act == 1) v = fmaxf(v, 0.f);
      Cp[(long)gm*sCm + gn] = v;
    }
    int gm2 = gm + 32;
    if (gm2 < M && gn < N) {
      float v = acc1[r];
      if (bias) v += bias[gn];
      if (act == 1) v = fmaxf(v, 0.f);
      Cp[(long)gm2*sCm + gn] = v;
    }
  }
}

// ---------------------------------------------------------------------------
// Patch embed: conv 4x4 stride 4, 1 input channel, output NHWC rows [B*Ho*Wo, C]
// ---------------------------------------------------------------------------
__global__ void k_patch_embed(const float* __restrict__ x, const float* __restrict__ w,
                              const float* __restrict__ b, float* __restrict__ out,
                              long n, int Hi, int Wi, int C)
{
  long i = blockIdx.x*(long)blockDim.x + threadIdx.x;
  if (i >= n) return;
  int Wo = Wi/4;
  int c = (int)(i % C); long p = i / C;
  int wo = (int)(p % Wo); long t = p / Wo;
  int Ho = Hi/4;
  int ho = (int)(t % Ho); int bb = (int)(t / Ho);
  float s = b[c];
  const float* xin = x + ((long)bb*Hi + ho*4)*Wi + wo*4;
  const float* wp  = w + (long)c*16;
#pragma unroll
  for (int ky = 0; ky < 4; ++ky)
#pragma unroll
    for (int kx = 0; kx < 4; ++kx)
      s += xin[ky*Wi + kx] * wp[ky*4 + kx];
  out[i] = s;
}

// LayerNorm over last dim; one block (128 threads) per row; in-place safe
__global__ void k_ln(const float* __restrict__ x, const float* __restrict__ g,
                     const float* __restrict__ b, float* __restrict__ out, int C)
{
  __shared__ float s1[128], s2[128];
  long row = blockIdx.x;
  const float* xr = x + row*(long)C;
  float a1 = 0.f, a2 = 0.f;
  for (int c = threadIdx.x; c < C; c += 128) { float v = xr[c]; a1 += v; a2 += v*v; }
  s1[threadIdx.x] = a1; s2[threadIdx.x] = a2;
  __syncthreads();
  for (int st = 64; st > 0; st >>= 1) {
    if ((int)threadIdx.x < st) { s1[threadIdx.x] += s1[threadIdx.x+st]; s2[threadIdx.x] += s2[threadIdx.x+st]; }
    __syncthreads();
  }
  float mean = s1[0] / C;
  float var  = s2[0] / C - mean*mean;
  float inv  = rsqrtf(var + 1e-5f);
  for (int c = threadIdx.x; c < C; c += 128)
    out[row*(long)C + c] = (xr[c] - mean) * inv * g[c] + b[c];
}

// rows [B*HW, rowStride] (cols colOff..colOff+D) -> NCHW [B, D, HW]
__global__ void k_rows_to_nchw(const float* __restrict__ rows, float* __restrict__ out,
                               long n, int HW, int rowStride, int colOff, int D)
{
  long i = blockIdx.x*(long)blockDim.x + threadIdx.x;
  if (i >= n) return;
  int l = (int)(i % HW); long t = i / HW;
  int d = (int)(t % D);  int bb = (int)(t / D);
  out[i] = rows[((long)bb*HW + l)*rowStride + colOff + d];
}

// depthwise 3x3 SAME conv + bias + SiLU on NCHW
__global__ void k_dwconv_silu(const float* __restrict__ x, const float* __restrict__ w,
                              const float* __restrict__ b, float* __restrict__ out,
                              long n, int H, int W, int D)
{
  long i = blockIdx.x*(long)blockDim.x + threadIdx.x;
  if (i >= n) return;
  int wx = (int)(i % W); long t = i / W;
  int hy = (int)(t % H); long bd = t / H;
  int d = (int)(bd % D);
  const float* xp = x + bd*(long)H*W;
  const float* wp = w + (long)d*9;
  float s = b[d];
  for (int ky = -1; ky <= 1; ++ky) {
    int yy = hy + ky; if (yy < 0 || yy >= H) continue;
    for (int kx = -1; kx <= 1; ++kx) {
      int xx = wx + kx; if (xx < 0 || xx >= W) continue;
      s += xp[yy*W + xx] * wp[(ky+1)*3 + (kx+1)];
    }
  }
  out[i] = siluf_(s);
}

// NCHW [B,D,H,W] -> xs [B,4,D,L] : hw, wh, flip(hw), flip(wh)
__global__ void k_build_xs(const float* __restrict__ xc, float* __restrict__ xs,
                           long n, int D, int H, int W)
{
  long i = blockIdx.x*(long)blockDim.x + threadIdx.x;
  if (i >= n) return;
  int L = H*W;
  int l = (int)(i % L); long t = i / L;
  int d = (int)(t % D); int bb = (int)(t / D);
  float v = xc[i];
  int h = l / W, wq = l % W;
  int lt = wq*H + h;
  long base   = (((long)bb*4)*D + d)*(long)L;
  long stride = (long)D*L;
  xs[base + l]                     = v;
  xs[base + stride + lt]           = v;
  xs[base + 2*stride + (L-1-l)]    = v;
  xs[base + 3*stride + (L-1-lt)]   = v;
}

// selective scan: thread per (b,k,d), N=16 states in registers
__global__ void k_scan(const float* __restrict__ xs, const float* __restrict__ dtlin,
                       const float* __restrict__ xdbl, const float* __restrict__ dt_b,
                       const float* __restrict__ A_logs, const float* __restrict__ Ds,
                       float* __restrict__ ys,
                       long n, int D, int L, int R, int Cdbl)
{
  const int N = 16;
  long i = blockIdx.x*(long)blockDim.x + threadIdx.x;
  if (i >= n) return;
  int d = (int)(i % D); long t = i / D;
  int k = (int)(t % 4); int bb = (int)(t / 4);
  long sbase = i * (long)L;  // i == ((bb*4+k)*D + d)
  const float* xsp = xs + sbase;
  const float* dtp = dtlin + sbase;
  float* ysp = ys + sbase;
  const float* Bsp = xdbl + (((long)(bb*4 + k))*Cdbl + R)*(long)L;
  const float* Csp = Bsp + (long)N*L;
  const float* Al  = A_logs + ((long)k*D + d)*N;
  float Ac[16];
#pragma unroll
  for (int nn = 0; nn < 16; ++nn) Ac[nn] = -__expf(Al[nn]);
  float dbias = dt_b[k*D + d];
  float Dv    = Ds[k*D + d];
  float h[16];
#pragma unroll
  for (int nn = 0; nn < 16; ++nn) h[nn] = 0.f;
  for (int l = 0; l < L; ++l) {
    float dt = softplusf_(dtp[l] + dbias);
    float u  = xsp[l];
    float du = dt * u;
    float y  = 0.f;
#pragma unroll
    for (int nn = 0; nn < 16; ++nn) {
      float dA = __expf(dt * Ac[nn]);
      h[nn] = h[nn]*dA + du*Bsp[(long)nn*L + l];
      y += h[nn]*Csp[(long)nn*L + l];
    }
    ysp[l] = y + Dv*u;
  }
}

// combine four direction outputs -> rows [B*L, D]
__global__ void k_combine(const float* __restrict__ ys, float* __restrict__ yrows,
                          long n, int D, int H, int W)
{
  long i = blockIdx.x*(long)blockDim.x + threadIdx.x;
  if (i >= n) return;
  int L = H*W;
  int d = (int)(i % D); long t = i / D;
  int l = (int)(t % L); int bb = (int)(t / L);
  int h = l / W, wq = l % W;
  int lt = wq*H + h;
  long base   = ((long)bb*4*D + d)*(long)L;
  long stride = (long)D*L;
  yrows[i] = ys[base + l] + ys[base + stride + lt]
           + ys[base + 2*stride + (L-1-l)] + ys[base + 3*stride + (L-1-lt)];
}

// y *= silu(z) where z = xz cols [D, 2D)
__global__ void k_gate(float* __restrict__ y, const float* __restrict__ xz, long n, int D)
{
  long i = blockIdx.x*(long)blockDim.x + threadIdx.x;
  if (i >= n) return;
  int d = (int)(i % D); long p = i / D;
  y[i] *= siluf_(xz[p*(long)(2*D) + D + d]);
}

__global__ void k_add(const float* __restrict__ a, const float* __restrict__ b,
                      float* __restrict__ c, long n)
{
  long i = blockIdx.x*(long)blockDim.x + threadIdx.x;
  if (i >= n) return;
  c[i] = a[i] + b[i];
}

__global__ void k_copy_cols(float* __restrict__ dst, int dstStride, int dstOff,
                            const float* __restrict__ src, int srcStride, int srcOff,
                            long n, int C)
{
  long i = blockIdx.x*(long)blockDim.x + threadIdx.x;
  if (i >= n) return;
  int c = (int)(i % C); long p = i / C;
  dst[p*(long)dstStride + dstOff + c] = src[p*(long)srcStride + srcOff + c];
}

// NCHW average pool to ps x ps
__global__ void k_avgpool(const float* __restrict__ x, float* __restrict__ out,
                          long n, int H, int W, int ps)
{
  long i = blockIdx.x*(long)blockDim.x + threadIdx.x;
  if (i >= n) return;
  int px = (int)(i % ps); long t = i / ps;
  int py = (int)(t % ps); long bc = t / ps;
  int bh = H/ps, bw = W/ps;
  const float* xp = x + bc*(long)H*W;
  float s = 0.f;
  for (int yy = 0; yy < bh; ++yy)
    for (int xx = 0; xx < bw; ++xx)
      s += xp[(py*bh + yy)*W + px*bw + xx];
  out[i] = s / (float)(bh*bw);
}

// pooled NCHW [B,C,S] -> [B,O,S] with relu, no bias
__global__ void k_conv1x1_relu2(const float* __restrict__ x, const float* __restrict__ w,
                                float* __restrict__ out, long n, int C, int O, int S)
{
  long i = blockIdx.x*(long)blockDim.x + threadIdx.x;
  if (i >= n) return;
  int sIdx = (int)(i % S); long t = i / S;
  int o = (int)(t % O); int bb = (int)(t / O);
  const float* xp = x + ((long)bb*C)*S + sIdx;
  const float* wp = w + (long)o*C;
  float s = 0.f;
  for (int c = 0; c < C; ++c) s += xp[(long)c*S] * wp[c];
  out[i] = fmaxf(s, 0.f);
}

// bilinear upsample [B,O,ps,ps] -> write into rows [B*H*W, Ct] at colOff (half-pixel)
__global__ void k_bilinear_concat(const float* __restrict__ sm, float* __restrict__ dst,
                                  long n, int O, int ps, int H, int W, int Ct, int colOff)
{
  long i = blockIdx.x*(long)blockDim.x + threadIdx.x;
  if (i >= n) return;
  int o = (int)(i % O); long t = i / O;
  int wx = (int)(t % W); t /= W;
  int hy = (int)(t % H); int bb = (int)(t / H);
  float sy = (hy + 0.5f) * (float)ps / (float)H - 0.5f;
  float sx = (wx + 0.5f) * (float)ps / (float)W - 0.5f;
  int y0 = (int)floorf(sy); float fy = sy - (float)y0;
  int x0 = (int)floorf(sx); float fx = sx - (float)x0;
  fy = fminf(fmaxf(fy, 0.f), 1.f); fx = fminf(fmaxf(fx, 0.f), 1.f);
  int y1 = min(max(y0+1,0),ps-1); y0 = min(max(y0,0),ps-1);
  int x1 = min(max(x0+1,0),ps-1); x0 = min(max(x0,0),ps-1);
  const float* sp = sm + ((long)bb*O + o)*(long)ps*ps;
  float v = (1.f-fy)*((1.f-fx)*sp[y0*ps+x0] + fx*sp[y0*ps+x1])
          +      fy *((1.f-fx)*sp[y1*ps+x0] + fx*sp[y1*ps+x1]);
  dst[((long)bb*H*W + hy*W + wx)*(long)Ct + colOff + o] = v;
}

__global__ void k_colmean(const float* __restrict__ rows, float* __restrict__ out,
                          long n, int HW, int C)
{
  long i = blockIdx.x*(long)blockDim.x + threadIdx.x;
  if (i >= n) return;
  int c = (int)(i % C); int bb = (int)(i / C);
  float s = 0.f;
  for (int l = 0; l < HW; ++l) s += rows[((long)bb*HW + l)*C + c];
  out[i] = s / (float)HW;
}

__global__ void k_matvec(const float* __restrict__ w, const float* __restrict__ bias,
                         const float* __restrict__ v, float* __restrict__ out,
                         long n, int O, int C)
{
  long i = blockIdx.x*(long)blockDim.x + threadIdx.x;
  if (i >= n) return;
  int o = (int)(i % O); int bb = (int)(i / O);
  float s = bias ? bias[o] : 0.f;
  const float* wp = w + (long)o*C;
  const float* vp = v + (long)bb*C;
  for (int c = 0; c < C; ++c) s += wp[c]*vp[c];
  out[i] = s;
}

__global__ void k_attn(const float* __restrict__ Q, const float* __restrict__ K,
                       const float* __restrict__ V, float* __restrict__ out,
                       long n, int Cr)
{
  long i = blockIdx.x*(long)blockDim.x + threadIdx.x;
  if (i >= n) return;
  int ii = (int)(i % Cr); int bb = (int)(i / Cr);
  float q = Q[(long)bb*Cr + ii];
  const float* Kp = K + (long)bb*Cr;
  const float* Vp = V + (long)bb*Cr;
  float mx = -1e30f;
  for (int j = 0; j < Cr; ++j) mx = fmaxf(mx, q*Kp[j]);
  float s = 0.f, acc = 0.f;
  for (int j = 0; j < Cr; ++j) { float e = __expf(q*Kp[j]-mx); s += e; acc += e*Vp[j]; }
  out[i] = acc / s;
}

__global__ void k_aux(const float* __restrict__ fuse, const float* __restrict__ p1,
                      const float* __restrict__ aw, const float* __restrict__ ab,
                      float* __restrict__ out, long n, int HW, int Cl)
{
  long i = blockIdx.x*(long)blockDim.x + threadIdx.x;
  if (i >= n) return;
  int l = (int)(i % HW); int bb = (int)(i / HW);
  const float* pp = p1 + ((long)bb*HW + l)*Cl;
  const float* fp = fuse + (long)bb*Cl;
  float s = ab[0];
  for (int c = 0; c < Cl; ++c) s += (fp[c] + pp[c]) * aw[c];
  out[i] = s;
}

// patch merge gather: rows [B*H*W, C] -> [B*(H/2)*(W/2), 4C]
__global__ void k_merge_gather(const float* __restrict__ x, float* __restrict__ out,
                               long n, int H, int W, int C)
{
  long i = blockIdx.x*(long)blockDim.x + threadIdx.x;
  if (i >= n) return;
  int C4 = 4*C, H2 = H/2, W2 = W/2;
  int c4 = (int)(i % C4); long t = i / C4;
  int w2 = (int)(t % W2); t /= W2;
  int h2 = (int)(t % H2); int bb = (int)(t / H2);
  int q = c4 / C, c = c4 % C;
  int dy = q & 1, dx = q >> 1;   // order: (0,0),(1,0),(0,1),(1,1)
  out[i] = x[(((long)bb*H + 2*h2 + dy)*W + 2*w2 + dx)*(long)C + c];
}

// ---------------------------------------------------------------------------
// Host orchestration
// ---------------------------------------------------------------------------
#define LAUNCH1D(kern, cnt, ...) do { long _n=(long)(cnt); if(_n>0){ \
  kern<<<dim3((unsigned)((_n+255)/256)), dim3(256), 0, stream>>>(__VA_ARGS__); } } while(0)

static void gemm(hipStream_t stream, const float* A, const float* B, float* C,
                 int M, int N, int K,
                 int sAm, int sAk, int sBk, int sBn, int sCm,
                 int nz, long sAz, long sBz, long sCz, int aZmod, int bZmod,
                 const float* bias, int act)
{
  dim3 grid((M+63)/64, (N+63)/64, nz);
  k_gemm<<<grid, dim3(256), 0, stream>>>(A, B, C, M, N, K,
      sAm, sAk, sBk, sBn, sCm, sAz, sBz, sCz, aZmod, bZmod, bias, act);
}

extern "C" void kernel_launch(void* const* d_in, const int* in_sizes, int n_in,
                              void* d_out, int out_size, void* d_ws, size_t ws_size,
                              hipStream_t stream)
{
  (void)in_sizes; (void)n_in; (void)out_size; (void)ws_size;
  int ci = 0;
  auto nx = [&]() { return (const float*)d_in[ci++]; };

  const float* x1 = nx();
  const float* x2 = nx();
  struct PE { const float *w,*b,*g,*beta; };
  PE pe1, pe2;
  pe1.w=nx(); pe1.b=nx(); pe1.g=nx(); pe1.beta=nx();
  pe2.w=nx(); pe2.b=nx(); pe2.g=nx(); pe2.beta=nx();
  struct DFEM { const float* ws[4]; const float* bot; };
  DFEM df1, df2;
  for (int i=0;i<4;++i) df1.ws[i]=nx();
  df1.bot=nx();
  for (int i=0;i<4;++i) df2.ws[i]=nx();
  df2.bot=nx();
  const float* proj1_w=nx(); const float* proj1_b=nx();
  const float* proj2_w=nx(); const float* proj2_b=nx();
  const float *qw=nx(),*qb=nx(),*kw=nx(),*kb=nx(),*vw=nx(),*vb=nx(),*ow=nx(),*ob=nx();
  const float* aux_w=nx(); const float* aux_b=nx();

  struct Blk { const float *ln_g,*ln_b,*in_proj,*conv_w,*conv_b,*x_proj,*dt_w,*dt_b,
               *A_logs,*Ds,*out_g,*out_b,*out_proj; };
  const int DIMS_[4]   = {96,192,384,768};
  const int DEPTHS_[4] = {2,2,9,2};
  Blk blocks[4][9];
  const float *mg_g[3], *mg_b[3], *mg_w[3];
  for (int s = 0; s < 4; ++s) {
    for (int d = 0; d < DEPTHS_[s]; ++d) {
      Blk& bk = blocks[s][d];
      bk.ln_g=nx(); bk.ln_b=nx(); bk.in_proj=nx(); bk.conv_w=nx(); bk.conv_b=nx();
      bk.x_proj=nx(); bk.dt_w=nx(); bk.dt_b=nx(); bk.A_logs=nx(); bk.Ds=nx();
      bk.out_g=nx(); bk.out_b=nx(); bk.out_proj=nx();
    }
    if (s < 3) { mg_g[s]=nx(); mg_b[s]=nx(); mg_w[s]=nx(); }
  }

  // ---- workspace (bump allocated, sized by stage-0 maxima) ----
  char* wsp = (char*)d_ws;
  size_t cur = 0;
  auto alloc = [&](size_t nfloats)->float* {
    float* p = (float*)(wsp + cur);
    cur += ((nfloats*sizeof(float) + 255)/256)*256;
    return p;
  };
  const int Bn = 2, H0 = 24, L0 = H0*H0, P0 = Bn*L0, C0 = 96;
  float* f1     = alloc((size_t)P0*C0);
  float* f2     = alloc((size_t)P0*C0);
  float* xA     = alloc((size_t)P0*C0);
  float* xB     = alloc((size_t)P0*C0);
  float* tbuf   = alloc((size_t)P0*C0);
  float* xz     = alloc((size_t)P0*384);
  float* xin    = alloc((size_t)P0*192);
  float* xc     = alloc((size_t)P0*192);
  float* xsb    = alloc((size_t)4*P0*192);
  float* xdbl   = alloc((size_t)2*4*38*L0);
  float* dtlin  = alloc((size_t)4*P0*192);
  float* ysb    = alloc((size_t)4*P0*192);
  float* yrows  = alloc((size_t)P0*192);
  float* blkout = alloc((size_t)P0*C0);
  float* mbuf   = alloc((size_t)P0*C0);
  float* nchw   = alloc((size_t)P0*C0);
  float* concat = alloc((size_t)P0*192);
  float* pooled = alloc((size_t)Bn*C0*36);
  float* pooledo= alloc((size_t)Bn*24*36);
  float* g1     = alloc((size_t)P0*C0);
  float* g2     = alloc((size_t)P0*C0);
  float* p1r    = alloc((size_t)P0*768);
  float* p2r    = alloc((size_t)P0*768);
  float* m1     = alloc((size_t)Bn*768);
  float* m2     = alloc((size_t)Bn*768);
  float* Qv     = alloc((size_t)Bn*192);
  float* Kv     = alloc((size_t)Bn*192);
  float* Vv     = alloc((size_t)Bn*192);
  float* attv   = alloc((size_t)Bn*192);
  float* fusev  = alloc((size_t)Bn*768);

  // ---- patch embeds + LN -> NHWC rows ----
  LAUNCH1D(k_patch_embed, (long)P0*C0, x1, pe1.w, pe1.b, f1, (long)P0*C0, 96, 96, C0);
  k_ln<<<dim3(P0), dim3(128), 0, stream>>>(f1, pe1.g, pe1.beta, f1, C0);
  LAUNCH1D(k_patch_embed, (long)P0*C0, x2, pe2.w, pe2.b, f2, (long)P0*C0, 96, 96, C0);
  k_ln<<<dim3(P0), dim3(128), 0, stream>>>(f2, pe2.g, pe2.beta, f2, C0);

  // ---- pyramid pooling path ----
  auto pyramid = [&](const float* frows, const DFEM& df, float* grows) {
    LAUNCH1D(k_rows_to_nchw, (long)Bn*C0*L0, frows, nchw, (long)Bn*C0*L0, L0, C0, 0, C0);
    LAUNCH1D(k_copy_cols, (long)P0*C0, concat, 192, 0, frows, C0, 0, (long)P0*C0, C0);
    const int pss[4] = {1,2,3,6};
    for (int i = 0; i < 4; ++i) {
      int ps = pss[i];
      LAUNCH1D(k_avgpool, (long)Bn*C0*ps*ps, nchw, pooled, (long)Bn*C0*ps*ps, H0, H0, ps);
      LAUNCH1D(k_conv1x1_relu2, (long)Bn*24*ps*ps, pooled, df.ws[i], pooledo,
               (long)Bn*24*ps*ps, C0, 24, ps*ps);
      LAUNCH1D(k_bilinear_concat, (long)Bn*L0*24, pooledo, concat,
               (long)Bn*L0*24, 24, ps, H0, H0, 192, C0 + 24*i);
    }
    gemm(stream, concat, df.bot, grows, P0, C0, 192,
         192, 1, 1, 192, C0, 1, 0, 0, 0, 0, 0, nullptr, 0);
  };
  pyramid(f1, df1, g1);
  pyramid(f2, df2, g2);

  // ---- proj to 768, cross-attention, aux head ----
  gemm(stream, g1, proj1_w, p1r, P0, 768, C0, C0, 1, 1, C0, 768, 1, 0,0,0, 0,0, proj1_b, 0);
  gemm(stream, g2, proj2_w, p2r, P0, 768, C0, C0, 1, 1, C0, 768, 1, 0,0,0, 0,0, proj2_b, 0);
  LAUNCH1D(k_colmean, (long)Bn*768, p1r, m1, (long)Bn*768, L0, 768);
  LAUNCH1D(k_colmean, (long)Bn*768, p2r, m2, (long)Bn*768, L0, 768);
  LAUNCH1D(k_matvec, (long)Bn*192, qw, qb, m1, Qv, (long)Bn*192, 192, 768);
  LAUNCH1D(k_matvec, (long)Bn*192, kw, kb, m2, Kv, (long)Bn*192, 192, 768);
  LAUNCH1D(k_matvec, (long)Bn*192, vw, vb, m2, Vv, (long)Bn*192, 192, 768);
  LAUNCH1D(k_attn, (long)Bn*192, Qv, Kv, Vv, attv, (long)Bn*192, 192);
  LAUNCH1D(k_matvec, (long)Bn*768, ow, ob, attv, fusev, (long)Bn*768, 768, 192);
  float* auxOut = (float*)d_out + (long)Bn*3*3*768;
  LAUNCH1D(k_aux, (long)Bn*L0, fusev, p1r, aux_w, aux_b, auxOut, (long)Bn*L0, L0, 768);

  // ---- main backbone ----
  float* x = xA; float* xalt = xB;
  LAUNCH1D(k_add, (long)P0*C0, f1, f2, x, (long)P0*C0);
  int H = 24, W = 24;
  for (int s = 0; s < 4; ++s) {
    int C = DIMS_[s];
    int L = H*W, P = Bn*L;
    int D = 2*C, R = (C+15)/16, Cdbl = R + 32;
    for (int bi = 0; bi < DEPTHS_[s]; ++bi) {
      const Blk& bk = blocks[s][bi];
      // t = ln(x)
      k_ln<<<dim3(P), dim3(128), 0, stream>>>(x, bk.ln_g, bk.ln_b, tbuf, C);
      // xz = t @ in_proj^T : [P, 2D]
      gemm(stream, tbuf, bk.in_proj, xz, P, 2*D, C,
           C, 1, 1, C, 2*D, 1, 0,0,0, 0,0, nullptr, 0);
      // xin (NCHW) = xz[:, :D]
      LAUNCH1D(k_rows_to_nchw, (long)Bn*D*L, xz, xin, (long)Bn*D*L, L, 2*D, 0, D);
      LAUNCH1D(k_dwconv_silu, (long)Bn*D*L, xin, bk.conv_w, bk.conv_b, xc,
               (long)Bn*D*L, H, W, D);
      LAUNCH1D(k_build_xs, (long)Bn*D*L, xc, xsb, (long)Bn*D*L, D, H, W);
      // x_dbl[z] = x_proj[k] @ xs[z] : batched z = b*4+k
      gemm(stream, bk.x_proj, xsb, xdbl, Cdbl, L, D,
           D, 1, L, 1, L, Bn*4, (long)Cdbl*D, (long)D*L, (long)Cdbl*L, 4, 0, nullptr, 0);
      // dts_lin[z] = dt_w[k] @ x_dbl[z][:R]
      gemm(stream, bk.dt_w, xdbl, dtlin, D, L, R,
           R, 1, L, 1, L, Bn*4, (long)D*R, (long)Cdbl*L, (long)D*L, 4, 0, nullptr, 0);
      // selective scan (softplus+bias folded in)
      LAUNCH1D(k_scan, (long)Bn*4*D, xsb, dtlin, xdbl, bk.dt_b, bk.A_logs, bk.Ds, ysb,
               (long)Bn*4*D, D, L, R, Cdbl);
      LAUNCH1D(k_combine, (long)P*D, ysb, yrows, (long)P*D, D, H, W);
      k_ln<<<dim3(P), dim3(128), 0, stream>>>(yrows, bk.out_g, bk.out_b, yrows, D);
      LAUNCH1D(k_gate, (long)P*D, yrows, xz, (long)P*D, D);
      gemm(stream, yrows, bk.out_proj, blkout, P, C, D,
           D, 1, 1, D, C, 1, 0,0,0, 0,0, nullptr, 0);
      LAUNCH1D(k_add, (long)P*C, x, blkout, x, (long)P*C);
    }
    if (s < 3) {
      int P2 = P/4;
      LAUNCH1D(k_merge_gather, (long)P2*4*C, x, mbuf, (long)P2*4*C, H, W, C);
      k_ln<<<dim3(P2), dim3(128), 0, stream>>>(mbuf, mg_g[s], mg_b[s], mbuf, 4*C);
      gemm(stream, mbuf, mg_w[s], xalt, P2, 2*C, 4*C,
           4*C, 1, 1, 4*C, 2*C, 1, 0,0,0, 0,0, nullptr, 0);
      float* tmp = x; x = xalt; xalt = tmp;
      H /= 2; W /= 2;
    }
  }

  // output: x (B,3,3,768) then aux (B,1,24,24) already written
  hipMemcpyAsync(d_out, x, (size_t)Bn*3*3*768*sizeof(float),
                 hipMemcpyDeviceToDevice, stream);
}
